// GNNNodeSelectionPolicy_57827439674230
// MI455X (gfx1250) — matrix operand order repaired and checked by reading, not compiled, not run on hardware
//
#include <hip/hip_runtime.h>
#include <hip/hip_bf16.h>
#include <stdint.h>

#define N_NODES 50000
#define N_EDGES 800000
#define IN_DIM  9
#define HIDDEN  256
#define N_CAND  4096

typedef __attribute__((ext_vector_type(16))) __bf16   v16bf;
typedef __attribute__((ext_vector_type(8)))  float    v8f;
typedef __attribute__((ext_vector_type(8)))  unsigned v8u;

static __device__ __forceinline__ unsigned short f2bf(float f) {
  // round-to-nearest-even fp32 -> bf16
  unsigned u = __float_as_uint(f);
  unsigned r = u + 0x7FFFu + ((u >> 16) & 1u);
  return (unsigned short)(r >> 16);
}

// ---------------- degree / normalization ----------------
__global__ void init_deg_kernel(float* __restrict__ deg) {
  int i = blockIdx.x * blockDim.x + threadIdx.x;
  if (i < N_NODES) deg[i] = 1.0f;  // self-loop
}

__global__ void edge_deg_kernel(const long long* __restrict__ ei, float* __restrict__ deg) {
  int i = blockIdx.x * blockDim.x + threadIdx.x;
  if (i < N_EDGES) atomicAdd(&deg[(int)ei[N_EDGES + i]], 1.0f);
}

__global__ void dinv_kernel(float* __restrict__ deg) {
  int i = blockIdx.x * blockDim.x + threadIdx.x;
  if (i < N_NODES) deg[i] = __frsqrt_rn(deg[i]);  // deg >= 1 always
}

// ---------------- weight transpose + bf16 convert ----------------
// in: K x N row-major fp32 (K=N=256). out: N x K row-major bf16.
__global__ void w_transpose_bf16_kernel(const float* __restrict__ in, unsigned short* __restrict__ out) {
  int idx = blockIdx.x * blockDim.x + threadIdx.x;
  if (idx < HIDDEN * HIDDEN) {
    int k = idx >> 8, n = idx & 255;
    out[n * HIDDEN + k] = f2bf(in[idx]);
  }
}

// ---------------- layer-1 GEMM (K=9, scalar; write-bound) ----------------
__global__ void gemm1_kernel(const float* __restrict__ x, const float* __restrict__ W1,
                             float* __restrict__ m) {
  __shared__ float xs[IN_DIM];
  int node = blockIdx.x;
  int t = threadIdx.x;
  if (t < IN_DIM) xs[t] = x[node * IN_DIM + t];
  __syncthreads();
  float s = 0.0f;
#pragma unroll
  for (int k = 0; k < IN_DIM; ++k) s += xs[k] * W1[k * HIDDEN + t];
  m[(size_t)node * HIDDEN + t] = s;
}

// ---------------- self-loop init: agg = m * dinv^2 ----------------
__global__ void selfloop_kernel(const float* __restrict__ m, const float* __restrict__ dinv,
                                float* __restrict__ agg) {
  size_t idx = (size_t)blockIdx.x * blockDim.x + threadIdx.x;
  if (idx < (size_t)N_NODES * HIDDEN) {
    int n = (int)(idx >> 8);
    float di = dinv[n];
    agg[idx] = m[idx] * di * di;
  }
}

// ---------------- edge scatter: one wave32 per edge ----------------
__global__ void scatter_kernel(const float* __restrict__ m, float* __restrict__ agg,
                               const long long* __restrict__ ei, const float* __restrict__ dinv) {
  int e = blockIdx.x * (blockDim.x >> 5) + (threadIdx.x >> 5);
  int lane = threadIdx.x & 31;
  if (e >= N_EDGES) return;
  int s = (int)ei[e];
  int d = (int)ei[N_EDGES + e];
  float w = dinv[s] * dinv[d];
  const float* mrow = m + (size_t)s * HIDDEN;
  float* arow = agg + (size_t)d * HIDDEN;
#pragma unroll
  for (int j = 0; j < HIDDEN / 32; ++j) {
    int f = j * 32 + lane;
    atomicAdd(&arow[f], mrow[f] * w);
  }
}

// ---------------- bias + ReLU -> bf16 activations ----------------
__global__ void bias_relu_bf16_kernel(const float* __restrict__ agg, const float* __restrict__ b,
                                      unsigned short* __restrict__ hb) {
  size_t idx = (size_t)blockIdx.x * blockDim.x + threadIdx.x;
  if (idx < (size_t)N_NODES * HIDDEN) {
    float v = agg[idx] + b[idx & 255];
    hb[idx] = f2bf(fmaxf(v, 0.0f));
  }
}

// ---------------- WMMA bf16 GEMM: C[M,256] = A[M,256] * Bt[256,256]^T ----------------
// A: bf16 row-major (K contiguous). Bt: bf16, row n = output column n, K contiguous.
// One wave per 16x16 output tile; K-loop of 8 steps of v_wmma_f32_16x16x32_bf16.
static __device__ __forceinline__ v8f wmma_tile(const unsigned short* __restrict__ Arow,
                                                const unsigned short* __restrict__ Brow,
                                                int half) {
  v8f c = {};
#pragma unroll
  for (int kk = 0; kk < HIDDEN; kk += 32) {
    unsigned au[8], bu[8];
#pragma unroll
    for (int v = 0; v < 8; ++v) {
      // A 16-bit layout: VGPR v holds K pair at ((v&3)*2 + (v>>2)*16 + 8*half)
      int ka = kk + ((v & 3) * 2) + ((v >> 2) * 16) + 8 * half;
      au[v] = *(const unsigned*)(Arow + ka);
      // B 16-bit layout: VGPR v holds K pair at ((v&3)*2 + (v>>2)*16 + 8*half)
      int kb = kk + ((v & 3) * 2) + ((v >> 2) * 16) + 8 * half;
      bu[v] = *(const unsigned*)(Brow + kb);
    }
    v8u av, bv;
#pragma unroll
    for (int v = 0; v < 8; ++v) { av[v] = au[v]; bv[v] = bu[v]; }
    c = __builtin_amdgcn_wmma_f32_16x16x32_bf16(
        false, __builtin_bit_cast(v16bf, av),
        false, __builtin_bit_cast(v16bf, bv),
        (short)0, c, false, false);
  }
  return c;
}

__global__ void wmma_gemm_kernel(const unsigned short* __restrict__ A,
                                 const unsigned short* __restrict__ Bt,
                                 float* __restrict__ C, int Mtiles) {
  int wid = blockIdx.x * (blockDim.x >> 5) + (threadIdx.x >> 5);
  int lane = threadIdx.x & 31;
  if (wid >= Mtiles * (HIDDEN / 16)) return;
  int tm = wid >> 4, tn = wid & 15;
  int half = lane >> 4, r = lane & 15;
  const unsigned short* Arow = A + (size_t)(tm * 16 + r) * HIDDEN;
  const unsigned short* Brow = Bt + (size_t)(tn * 16 + r) * HIDDEN;
  v8f c = wmma_tile(Arow, Brow, half);
  // C/D layout: VGPR v -> row (v + 8*half), lanes&15 -> column
  float* Crow = C + (size_t)tm * 16 * HIDDEN + tn * 16;
#pragma unroll
  for (int v = 0; v < 8; ++v)
    Crow[(size_t)(v + 8 * half) * HIDDEN + r] = c[v];
}

// Head GEMM with bias + tanh epilogue: th = tanh(cand @ Wh1 + bh1)
__global__ void wmma_head_kernel(const unsigned short* __restrict__ A,
                                 const unsigned short* __restrict__ Bt,
                                 const float* __restrict__ bh1,
                                 float* __restrict__ th, int Mtiles) {
  int wid = blockIdx.x * (blockDim.x >> 5) + (threadIdx.x >> 5);
  int lane = threadIdx.x & 31;
  if (wid >= Mtiles * (HIDDEN / 16)) return;
  int tm = wid >> 4, tn = wid & 15;
  int half = lane >> 4, r = lane & 15;
  const unsigned short* Arow = A + (size_t)(tm * 16 + r) * HIDDEN;
  const unsigned short* Brow = Bt + (size_t)(tn * 16 + r) * HIDDEN;
  v8f c = wmma_tile(Arow, Brow, half);
  float bias = bh1[tn * 16 + r];
  float* Crow = th + (size_t)tm * 16 * HIDDEN + tn * 16;
#pragma unroll
  for (int v = 0; v < 8; ++v)
    Crow[(size_t)(v + 8 * half) * HIDDEN + r] = tanhf(c[v] + bias);
}

// ---------------- candidate gather (bf16 rows) ----------------
__global__ void gather_cand_kernel(const unsigned short* __restrict__ hb,
                                   const long long* __restrict__ cand,
                                   unsigned short* __restrict__ cb) {
  int b = blockIdx.x, t = threadIdx.x;
  cb[(size_t)b * HIDDEN + t] = hb[(size_t)((int)cand[b]) * HIDDEN + t];
}

// ---------------- final dot: scores[k] = th[k,:] . Wh2 + bh2 ----------------
__global__ void head_final_kernel(const float* __restrict__ th, const float* __restrict__ Wh2,
                                  const float* __restrict__ bh2, float* __restrict__ out) {
  __shared__ float red[HIDDEN];
  int b = blockIdx.x, t = threadIdx.x;
  red[t] = th[(size_t)b * HIDDEN + t] * Wh2[t];
  __syncthreads();
#pragma unroll
  for (int s = HIDDEN / 2; s > 0; s >>= 1) {
    if (t < s) red[t] += red[t + s];
    __syncthreads();
  }
  if (t == 0) out[b] = red[0] + bh2[0];
}

extern "C" void kernel_launch(void* const* d_in, const int* in_sizes, int n_in,
                              void* d_out, int out_size, void* d_ws, size_t ws_size,
                              hipStream_t stream) {
  const float*     x    = (const float*)d_in[0];
  const float*     W1   = (const float*)d_in[1];
  const float*     b1   = (const float*)d_in[2];
  const float*     W2   = (const float*)d_in[3];
  const float*     b2   = (const float*)d_in[4];
  const float*     Wh1  = (const float*)d_in[5];
  const float*     bh1  = (const float*)d_in[6];
  const float*     Wh2  = (const float*)d_in[7];
  const float*     bh2  = (const float*)d_in[8];
  const long long* ei   = (const long long*)d_in[9];   // int64 [2, E]
  const long long* cand = (const long long*)d_in[10];  // int64 [K]
  float* scores = (float*)d_out;

  // workspace carve-out
  uintptr_t p = (uintptr_t)d_ws;
  auto carve = [&](size_t bytes) -> void* {
    uintptr_t q = (p + 255) & ~(uintptr_t)255;
    p = q + bytes;
    return (void*)q;
  };
  float*          dinv = (float*)carve((size_t)N_NODES * 4);
  float*          m    = (float*)carve((size_t)N_NODES * HIDDEN * 4);
  float*          agg  = (float*)carve((size_t)N_NODES * HIDDEN * 4);
  unsigned short* hb   = (unsigned short*)carve((size_t)N_NODES * HIDDEN * 2);
  unsigned short* W2t  = (unsigned short*)carve((size_t)HIDDEN * HIDDEN * 2);
  unsigned short* Wh1t = (unsigned short*)carve((size_t)HIDDEN * HIDDEN * 2);
  unsigned short* cb   = (unsigned short*)carve((size_t)N_CAND * HIDDEN * 2);
  float*          th   = (float*)carve((size_t)N_CAND * HIDDEN * 4);
  (void)ws_size; (void)in_sizes; (void)n_in; (void)out_size;

  const size_t NH = (size_t)N_NODES * HIDDEN;
  const int nhBlocks = (int)((NH + 255) / 256);

  // normalization
  init_deg_kernel<<<(N_NODES + 255) / 256, 256, 0, stream>>>(dinv);
  edge_deg_kernel<<<(N_EDGES + 255) / 256, 256, 0, stream>>>(ei, dinv);
  dinv_kernel<<<(N_NODES + 255) / 256, 256, 0, stream>>>(dinv);

  // bf16 transposed weights
  w_transpose_bf16_kernel<<<(HIDDEN * HIDDEN + 255) / 256, 256, 0, stream>>>(W2, W2t);
  w_transpose_bf16_kernel<<<(HIDDEN * HIDDEN + 255) / 256, 256, 0, stream>>>(Wh1, Wh1t);

  // ----- layer 1 -----
  gemm1_kernel<<<N_NODES, HIDDEN, 0, stream>>>(x, W1, m);
  selfloop_kernel<<<nhBlocks, 256, 0, stream>>>(m, dinv, agg);
  scatter_kernel<<<(N_EDGES + 7) / 8, 256, 0, stream>>>(m, agg, ei, dinv);
  bias_relu_bf16_kernel<<<nhBlocks, 256, 0, stream>>>(agg, b1, hb);

  // ----- layer 2 (WMMA bf16 GEMM) -----
  {
    int Mtiles = N_NODES / 16;           // 3125, exact
    int waves = Mtiles * (HIDDEN / 16);  // 50000
    wmma_gemm_kernel<<<(waves + 7) / 8, 256, 0, stream>>>(hb, W2t, m, Mtiles);
  }
  selfloop_kernel<<<nhBlocks, 256, 0, stream>>>(m, dinv, agg);
  scatter_kernel<<<(N_EDGES + 7) / 8, 256, 0, stream>>>(m, agg, ei, dinv);
  bias_relu_bf16_kernel<<<nhBlocks, 256, 0, stream>>>(agg, b2, hb);

  // ----- score head -----
  gather_cand_kernel<<<N_CAND, HIDDEN, 0, stream>>>(hb, cand, cb);
  {
    int Mtiles = N_CAND / 16;            // 256, exact
    int waves = Mtiles * (HIDDEN / 16);  // 4096
    wmma_head_kernel<<<(waves + 7) / 8, 256, 0, stream>>>(cb, Wh1t, bh1, th, Mtiles);
  }
  head_final_kernel<<<N_CAND, HIDDEN, 0, stream>>>(th, Wh2, bh2, scores);
}
